// FeatureResponseGenerator_53498112639194
// MI455X (gfx1250) — matrix operand order repaired and checked by reading, not compiled, not run on hardware
//
#include <hip/hip_runtime.h>

typedef _Float16 v16h __attribute__((ext_vector_type(16)));
typedef _Float16 v8h  __attribute__((ext_vector_type(8)));
typedef float    v8f  __attribute__((ext_vector_type(8)));
typedef float    v4f  __attribute__((ext_vector_type(4)));

#define B_  4
#define C_  128
#define H_  96
#define W_  128
#define HW_ (H_ * W_)          // 12288
#define S_  1024

// resp = exp(SCALE*(0.5*corr + 0.5 - THRESH)) = exp(10*corr - 9.99)
#define COEF_A 10.0f
#define COEF_B (-9.99f)

// ---------------------------------------------------------------------------
// workspace layout:
//   [0, B*S*C*2)            : A16  [B][S][C]  _Float16 (gathered query vectors)
//   [B*S*C*2, +B*S*4)       : sums [B][S]     float
// ---------------------------------------------------------------------------

__global__ void zero_sums_kernel(float* __restrict__ sums) {
    int i = blockIdx.x * blockDim.x + threadIdx.x;
    if (i < B_ * S_) sums[i] = 0.0f;
}

// Gather sampled query vectors and convert to f16: A16[b][s][c] = src[b][c][loc[b,s]]
__global__ void gather_a16_kernel(const float* __restrict__ src,
                                  const int* __restrict__ loc,
                                  _Float16* __restrict__ a16) {
    int idx = blockIdx.x * blockDim.x + threadIdx.x;   // over B*S*C
    int c  = idx & (C_ - 1);
    int bs = idx >> 7;                                 // b*S + s  (C=128)
    int b  = bs >> 10;                                 // S=1024
    int l  = loc[bs];
    a16[idx] = (_Float16)src[((size_t)(b * C_ + c)) * HW_ + l];
}

// One wave computes a 32(S) x 16(HW) tile with K=C=128 via 4x2 chained
// v_wmma_f32_16x16x32_f16. 4 waves per block cover 32 x 64.
__global__ __launch_bounds__(128)
void corr_gemm_kernel(const _Float16* __restrict__ A16,  // [B][S][C] f16
                      const float*    __restrict__ T,    // [B][C][HW] f32
                      const float*    __restrict__ mask, // [B][HW]
                      float*          __restrict__ out,  // [B][S][HW]
                      float*          __restrict__ sums) // [B][S]
{
    const int wave = threadIdx.x >> 5;
    const int lane = threadIdx.x & 31;
    const int b    = blockIdx.z;
    const int s0   = blockIdx.y * 32;
    const int hw0  = blockIdx.x * 64 + wave * 16;

    const int grp = lane >> 4;     // 0 or 1 (half-wave)
    const int m   = lane & 15;     // A row within 16-row tile

    const _Float16* arow0 = A16 + ((size_t)(b * S_ + s0 + m)) * C_;       // rows s0..s0+15
    const _Float16* arow1 = arow0 + 16 * C_;                              // rows s0+16..s0+31
    const float*    tcol  = T + ((size_t)b * C_) * HW_ + hw0 + m;         // lane indexes N

    v8f acc0 = {};
    v8f acc1 = {};

#pragma unroll
    for (int kk = 0; kk < 4; ++kk) {
        const int kbase = kk * 32;

        // B tile (32x16): lane = N = hw0 + (lane&15), elements = K in same order as A (f32 -> f16)
        const float* tb = tcol + (size_t)kbase * HW_;
        v16h bv;
#pragma unroll
        for (int i = 0; i < 16; ++i) bv[i] = (_Float16)tb[(size_t)(grp * 8 + (i & 7) + (i >> 3) * 16) * HW_];

        // A tile: lane (grp,m); elements 0..7 -> K = kbase + grp*8 + e,
        //                       elements 8..15 -> K = kbase + 16 + grp*8 + e
        const int k1 = kbase + grp * 8;
        const int k2 = kbase + 16 + grp * 8;
        v8h a0lo = *(const v8h*)(arow0 + k1);
        v8h a0hi = *(const v8h*)(arow0 + k2);
        v8h a1lo = *(const v8h*)(arow1 + k1);
        v8h a1hi = *(const v8h*)(arow1 + k2);
        v16h av0 = __builtin_shufflevector(a0lo, a0hi, 0,1,2,3,4,5,6,7,8,9,10,11,12,13,14,15);
        v16h av1 = __builtin_shufflevector(a1lo, a1hi, 0,1,2,3,4,5,6,7,8,9,10,11,12,13,14,15);

        acc0 = __builtin_amdgcn_wmma_f32_16x16x32_f16(false, av0, false, bv,
                                                      (short)0, acc0, false, false);
        acc1 = __builtin_amdgcn_wmma_f32_16x16x32_f16(false, av1, false, bv,
                                                      (short)0, acc1, false, false);
    }

    // Epilogue: D layout -> lane holds N = lane&15, rows M = r + 8*grp
    const int n  = lane & 15;
    const int hw = hw0 + n;
    const float mval = mask[(size_t)b * HW_ + hw];

    float rs0[8], rs1[8];
#pragma unroll
    for (int r = 0; r < 8; ++r) {
        float e0 = __expf(COEF_A * acc0[r] + COEF_B) * mval;
        float e1 = __expf(COEF_A * acc1[r] + COEF_B) * mval;
        const int srow0 = s0 + r + 8 * grp;
        const int srow1 = srow0 + 16;
        out[((size_t)(b * S_ + srow0)) * HW_ + hw] = e0;
        out[((size_t)(b * S_ + srow1)) * HW_ + hw] = e1;
        rs0[r] = e0;
        rs1[r] = e1;
    }

    // Reduce each row's 16 N-values across the half-wave, then one atomic per row.
#pragma unroll
    for (int r = 0; r < 8; ++r) {
        float v0 = rs0[r];
        float v1 = rs1[r];
#pragma unroll
        for (int off = 1; off < 16; off <<= 1) {
            v0 += __shfl_xor(v0, off, 32);
            v1 += __shfl_xor(v1, off, 32);
        }
        if (n == 0) {
            atomicAdd(&sums[b * S_ + s0 + r + 8 * grp], v0);
            atomicAdd(&sums[b * S_ + s0 + 16 + r + 8 * grp], v1);
        }
    }
}

// Divide every element by its row sum (float4 per thread for bandwidth).
__global__ void normalize_kernel(float* __restrict__ out,
                                 const float* __restrict__ sums) {
    size_t i4   = (size_t)blockIdx.x * blockDim.x + threadIdx.x;
    size_t base = i4 * 4;
    int row = (int)(base / HW_);   // HW_ % 4 == 0, all 4 lanes share a row
    float inv = 1.0f / sums[row];
    v4f v = *(v4f*)(out + base);
    v.x *= inv; v.y *= inv; v.z *= inv; v.w *= inv;
    *(v4f*)(out + base) = v;
}

extern "C" void kernel_launch(void* const* d_in, const int* in_sizes, int n_in,
                              void* d_out, int out_size, void* d_ws, size_t ws_size,
                              hipStream_t stream) {
    const float* src  = (const float*)d_in[0];   // [B,C,H,W]
    const float* tgt  = (const float*)d_in[1];   // [B,C,H,W]
    const int*   loc  = (const int*)d_in[2];     // [B,S,1] (int32)
    const float* mask = (const float*)d_in[3];   // [B,1,H,W]
    float* out = (float*)d_out;                  // [B,S,H,W]

    _Float16* a16 = (_Float16*)d_ws;
    float* sums = (float*)((char*)d_ws + (size_t)B_ * S_ * C_ * sizeof(_Float16));

    zero_sums_kernel<<<(B_ * S_ + 255) / 256, 256, 0, stream>>>(sums);
    gather_a16_kernel<<<(B_ * S_ * C_) / 256, 256, 0, stream>>>(src, loc, a16);

    dim3 grid(HW_ / 64, S_ / 32, B_);
    corr_gemm_kernel<<<grid, 128, 0, stream>>>(a16, tgt, mask, out, sums);

    normalize_kernel<<<((size_t)B_ * S_ * HW_ / 4 + 255) / 256, 256, 0, stream>>>(out, sums);
}
